// EGNN_86268713107937
// MI455X (gfx1250) — matrix operand order repaired and checked by reading, not compiled, not run on hardware
//
#include <hip/hip_runtime.h>
#include <hip/hip_bf16.h>
#include <cstdint>
#include <cstddef>

typedef __attribute__((ext_vector_type(16))) _Float16 v16h;
typedef __attribute__((ext_vector_type(8)))  _Float16 v8h;
typedef __attribute__((ext_vector_type(8)))  float    v8f;
typedef __attribute__((ext_vector_type(4)))  float    v4f;

#define EPS8 1e-8f

__device__ __forceinline__ float silu_f(float v) { return v / (1.f + __expf(-v)); }

__device__ __forceinline__ v16h cat8(v8h a, v8h b) {
  return __builtin_shufflevector(a, b, 0,1,2,3,4,5,6,7,8,9,10,11,12,13,14,15);
}

__device__ __forceinline__ v8f wmma_f16(v16h a, v16h b, v8f c) {
  return __builtin_amdgcn_wmma_f32_16x16x32_f16(false, a, false, b, (short)0, c, false, false);
}

// 8 consecutive f32 -> 8 f16 (two 16B vector loads)
__device__ __forceinline__ v8h cvt8_raw(const float* p) {
  const v4f* q = (const v4f*)p;
  v4f x0 = q[0], x1 = q[1];
  v8h r;
#pragma unroll
  for (int i = 0; i < 4; ++i) { r[i] = (_Float16)x0[i]; r[4+i] = (_Float16)x1[i]; }
  return r;
}

// 8 consecutive f32 with fused instance-norm scale/shift + silu -> f16
__device__ __forceinline__ v8h cvt8_ns(const float* p, const float* sc, const float* sh) {
  const v4f* q = (const v4f*)p;
  const v4f* s = (const v4f*)sc;
  const v4f* t = (const v4f*)sh;
  v4f x0 = q[0], x1 = q[1], s0 = s[0], s1 = s[1], t0 = t[0], t1 = t[1];
  v8h r;
#pragma unroll
  for (int i = 0; i < 4; ++i) {
    r[i]   = (_Float16)silu_f(x0[i]*s0[i]+t0[i]);
    r[4+i] = (_Float16)silu_f(x1[i]*s1[i]+t1[i]);
  }
  return r;
}

// A fragment (16-bit A 16x32 layout): lane supplies row m; halves 0-7 are
// K = k0+hi*8 .. +7, halves 8-15 are K = k0+16+hi*8 .. +7 (contiguous groups).
__device__ __forceinline__ v16h frag_a_raw(const float* rowp, int k0, int hi) {
  return cat8(cvt8_raw(rowp + k0 + hi*8), cvt8_raw(rowp + k0 + 16 + hi*8));
}
__device__ __forceinline__ v16h frag_a_ns(const float* rowp, int k0, int hi,
                                          const float* sc, const float* sh) {
  int o0 = k0 + hi*8, o1 = k0 + 16 + hi*8;
  return cat8(cvt8_ns(rowp + o0, sc + o0, sh + o0),
              cvt8_ns(rowp + o1, sc + o1, sh + o1));
}

// B fragment from fragment-major swizzled LDS: 16 contiguous halves per lane.
__device__ __forceinline__ v16h frag_b(const _Float16* Bs, int fid, int lane) {
  const v8h* p = (const v8h*)(Bs + (((size_t)fid * 32 + lane) << 4));
  return cat8(p[0], p[1]);
}

// Stage W (K x ncols f32, row-major, ncols = NT*16) into fragment-major f16:
// Bs[((kc*NT + n)*32 + lane)*16 + i] = W[(kc*32 + (i>>3)*16 + hi*8 + (i&7))*ncols + n*16 + m]
__device__ __forceinline__ void load_B_swz(_Float16* Bs, const float* __restrict__ W,
                                           int K, int Kpad, int NT_) {
  int ncols = NT_ * 16;
  int total = (Kpad >> 5) * NT_ * 512;
  for (int idx = threadIdx.x; idx < total; idx += blockDim.x) {
    int i    = idx & 15;
    int lane = (idx >> 4) & 31;
    int fid  = idx >> 9;
    int n    = fid % NT_;
    int kc   = fid / NT_;
    int k = kc*32 + (i >> 3)*16 + (lane >> 4)*8 + (i & 7);
    int j = n*16 + (lane & 15);
    Bs[idx] = (k < K) ? (_Float16)W[(size_t)k * ncols + j] : (_Float16)0.f;
  }
}

// C/D f32 tile store: lane (m,hi), VGPR r -> row = hi*8+r, col = n*16+m.
// STATS: accumulate per-column sum/sumsq of stored (pre-activation) values into
// block LDS (bls[0:64]=sum, bls[64:128]=sumsq) via shfl + LDS atomics.
template<int NT, int MODE_OUT, bool STATS>
__device__ __forceinline__ void store_tile(float* __restrict__ out, int ldo, int row0,
                                           int rows, const float* __restrict__ bias,
                                           const v8f* acc, int m, int hi, float* bls) {
  if (row0 + 16 <= rows) {
#pragma unroll
    for (int n = 0; n < NT; ++n) {
      float b = bias[n*16 + m];
      float s = 0.f, q = 0.f;
#pragma unroll
      for (int r = 0; r < 8; ++r) {
        float v = acc[n][r] + b;
        if (STATS) { s += v; q += v * v; }
        if (MODE_OUT == 1) v = silu_f(v);
        size_t idx = (size_t)(row0 + hi*8 + r) * ldo + n*16 + m;
        if (MODE_OUT == 2) out[idx] += v; else out[idx] = v;
      }
      if (STATS) {
        s += __shfl_xor(s, 16);
        q += __shfl_xor(q, 16);
        if (hi == 0) {
          atomicAdd(&bls[n*16 + m], s);
          atomicAdd(&bls[64 + n*16 + m], q);
        }
      }
    }
  } else {
#pragma unroll
    for (int n = 0; n < NT; ++n) {
      float b = bias[n*16 + m];
      float s = 0.f, q = 0.f;
#pragma unroll
      for (int r = 0; r < 8; ++r) {
        int row = row0 + hi*8 + r;
        if (row < rows) {
          float v = acc[n][r] + b;
          if (STATS) { s += v; q += v * v; }
          if (MODE_OUT == 1) v = silu_f(v);
          size_t idx = (size_t)row * ldo + n*16 + m;
          if (MODE_OUT == 2) out[idx] += v; else out[idx] = v;
        }
      }
      if (STATS) {
        s += __shfl_xor(s, 16);
        q += __shfl_xor(q, 16);
        if (hi == 0) {
          atomicAdd(&bls[n*16 + m], s);
          atomicAdd(&bls[64 + n*16 + m], q);
        }
      }
    }
  }
}

// ---------------------------------------------------------------------------
// Geometry / setup kernels
// ---------------------------------------------------------------------------
__global__ void k_node_accum(const float* __restrict__ pc, const int* __restrict__ tag,
                             float* __restrict__ barysum, float* __restrict__ cnt, int N)
{
  __shared__ float ls[256 * 4];
  for (int i = threadIdx.x; i < 1024; i += blockDim.x) ls[i] = 0.f;
  __syncthreads();
  int i = blockIdx.x * blockDim.x + threadIdx.x;
  if (i < N) {
    int t = tag[i] & 255;
    atomicAdd(&ls[t*4+0], pc[3*i+0]);
    atomicAdd(&ls[t*4+1], pc[3*i+1]);
    atomicAdd(&ls[t*4+2], pc[3*i+2]);
    atomicAdd(&ls[t*4+3], 1.f);
  }
  __syncthreads();
  for (int t = threadIdx.x; t < 256; t += blockDim.x) {
    if (ls[t*4+3] != 0.f) {
      atomicAdd(&barysum[t*3+0], ls[t*4+0]);
      atomicAdd(&barysum[t*3+1], ls[t*4+1]);
      atomicAdd(&barysum[t*3+2], ls[t*4+2]);
      atomicAdd(&cnt[t], ls[t*4+3]);
    }
  }
}

__global__ void k_edge_geom(const float* __restrict__ pc, const int* __restrict__ src,
                            const int* __restrict__ dst, float* __restrict__ dir,
                            float* __restrict__ dist, float* __restrict__ deg,
                            float* __restrict__ densSum, int E)
{
  int e = blockIdx.x * blockDim.x + threadIdx.x;
  if (e >= E) return;
  int s = src[e], d = dst[e];
  float dx = pc[3*s+0] - pc[3*d+0];
  float dy = pc[3*s+1] - pc[3*d+1];
  float dz = pc[3*s+2] - pc[3*d+2];
  float dd = sqrtf(dx*dx + dy*dy + dz*dz);
  float inv = 1.f / (dd + EPS8);
  dir[3*e+0] = dx * inv;
  dir[3*e+1] = dy * inv;
  dir[3*e+2] = dz * inv;
  dist[e] = dd;
  atomicAdd(&deg[s], 1.f);
  atomicAdd(&densSum[s], dd);
}

__global__ void k_node_init(const float* __restrict__ pc, const int* __restrict__ tag,
                            const float* __restrict__ barysum, const float* __restrict__ cnt,
                            const float* __restrict__ deg, const float* __restrict__ densSum,
                            const float* __restrict__ w_emb, const float* __restrict__ b_emb,
                            float* __restrict__ x, float* __restrict__ h, int N)
{
  int i = blockIdx.x * blockDim.x + threadIdx.x;
  if (i >= N) return;
  int t = tag[i] & 255;
  float c = cnt[t];
  float px = pc[3*i+0], py = pc[3*i+1], pz = pc[3*i+2];
  x[3*i+0] = px - barysum[t*3+0] / c;
  x[3*i+1] = py - barysum[t*3+1] / c;
  x[3*i+2] = pz - barysum[t*3+2] / c;
  float loc  = sqrtf(px*px + py*py + pz*pz);
  float dens = densSum[i] / fmaxf(deg[i], 1.f);
  v4f* hr = (v4f*)(h + (size_t)i * 64);
  const v4f* w0 = (const v4f*)w_emb;
  const v4f* w1 = (const v4f*)(w_emb + 64);
  const v4f* bb = (const v4f*)b_emb;
#pragma unroll
  for (int j = 0; j < 16; ++j) hr[j] = loc * w0[j] + dens * w1[j] + bb[j];
}

__global__ void k_edge_attr(const float* __restrict__ dir, const int* __restrict__ src,
                            const int* __restrict__ dst, float* __restrict__ ang, int E)
{
  int e = blockIdx.x * blockDim.x + threadIdx.x;
  if (e >= E) return;
  int s = src[e], d = dst[e];   // faithful to source: node-id indexing of per-edge dir
  float a = dir[3*s+0]*dir[3*d+0] + dir[3*s+1]*dir[3*d+1] + dir[3*s+2]*dir[3*d+2];
  a = fminf(fmaxf(a, -1.f + EPS8), 1.f - EPS8);
  ang[e] = acosf(a);
}

// ---------------------------------------------------------------------------
// Edge GEMM 1: t1 = [h[src] | h[dst] | radial | dist | angle] @ ew1 + eb1
// K = 131 (pad 160). Persistent tile loop; fused column stats.
// ---------------------------------------------------------------------------
__global__ void k_edge_gemm1(const float* __restrict__ h, const float* __restrict__ x,
                             const float* __restrict__ dist, const float* __restrict__ ang,
                             const int* __restrict__ src, const int* __restrict__ dst,
                             const float* __restrict__ W, const float* __restrict__ bias,
                             float* __restrict__ out, float* __restrict__ sums, int E)
{
  __shared__ _Float16 Bs[5 * 4 * 512];   // 20 KB
  __shared__ float bls[128];
  load_B_swz(Bs, W, 131, 160, 4);
  for (int j = threadIdx.x; j < 128; j += blockDim.x) bls[j] = 0.f;
  __syncthreads();
  const int lane = threadIdx.x & 31, wave = threadIdx.x >> 5;
  const int m = lane & 15, hi = lane >> 4;
  const int numTiles = (E + 15) >> 4;
  const int stride = gridDim.x * 8;

  for (int tile = blockIdx.x * 8 + wave; tile < numTiles; tile += stride) {
    const int row0 = tile * 16;
    const int e = min(row0 + m, E - 1);
    const int s = src[e], d = dst[e];
    const float* hs = h + (size_t)s * 64;
    const float* hd = h + (size_t)d * 64;
    float vdx = x[3*s+0]-x[3*d+0], vdy = x[3*s+1]-x[3*d+1], vdz = x[3*s+2]-x[3*d+2];
    float radial = vdx*vdx + vdy*vdy + vdz*vdz;

    v8f acc[4] = {};
    v16h a;
#pragma unroll
    for (int kc = 0; kc < 5; ++kc) {
      if      (kc < 2) a = frag_a_raw(hs, kc*32, hi);
      else if (kc < 4) a = frag_a_raw(hd, (kc-2)*32, hi);
      else {
        v16h z;
#pragma unroll
        for (int i = 0; i < 16; ++i) z[i] = (_Float16)0.f;
        z[0] = (_Float16)(hi ? 0.f : radial);
        z[1] = (_Float16)(hi ? 0.f : dist[e]);
        z[2] = (_Float16)(hi ? 0.f : ang[e]);
        a = z;
      }
#pragma unroll
      for (int n = 0; n < 4; ++n)
        acc[n] = wmma_f16(a, frag_b(Bs, kc*4 + n, lane), acc[n]);
    }
    store_tile<4, 0, true>(out, 64, row0, E, bias, acc, m, hi, bls);
  }
  __syncthreads();
  for (int j = threadIdx.x; j < 128; j += blockDim.x) atomicAdd(&sums[j], bls[j]);
}

// ---------------------------------------------------------------------------
// Generic rows x 64 @ 64 x 64 GEMM, A direct from global, B held in registers.
// MODE_IN : 0 = raw A, 1 = A = silu(in*sc + sh)
// MODE_OUT: 0 = store, 1 = store silu, 2 = accumulate into out
// ---------------------------------------------------------------------------
template<int MODE_IN, int MODE_OUT, bool STATS>
__global__ void k_gemm64(const float* __restrict__ in, const float* __restrict__ sc,
                         const float* __restrict__ sh, const float* __restrict__ W,
                         const float* __restrict__ bias, float* __restrict__ out,
                         float* __restrict__ sums, int rows)
{
  __shared__ _Float16 Bs[2 * 4 * 512];   // 8 KB
  __shared__ float scs[64], shs[64];
  __shared__ float bls[128];
  load_B_swz(Bs, W, 64, 64, 4);
  if (MODE_IN == 1)
    for (int j = threadIdx.x; j < 64; j += blockDim.x) { scs[j] = sc[j]; shs[j] = sh[j]; }
  if (STATS)
    for (int j = threadIdx.x; j < 128; j += blockDim.x) bls[j] = 0.f;
  __syncthreads();
  const int lane = threadIdx.x & 31, wave = threadIdx.x >> 5;
  const int m = lane & 15, hi = lane >> 4;
  v16h bf[8];
#pragma unroll
  for (int f = 0; f < 8; ++f) bf[f] = frag_b(Bs, f, lane);  // B lives in 64 VGPRs

  const int numTiles = (rows + 15) >> 4;
  const int stride = gridDim.x * 8;
  for (int tile = blockIdx.x * 8 + wave; tile < numTiles; tile += stride) {
    const int row0 = tile * 16;
    const float* rp = in + (size_t)min(row0 + m, rows - 1) * 64;
    v8f acc[4] = {};
#pragma unroll
    for (int kc = 0; kc < 2; ++kc) {
      v16h a = (MODE_IN == 1) ? frag_a_ns(rp, kc*32, hi, scs, shs)
                              : frag_a_raw(rp, kc*32, hi);
#pragma unroll
      for (int n = 0; n < 4; ++n)
        acc[n] = wmma_f16(a, bf[kc*4 + n], acc[n]);
    }
    store_tile<4, MODE_OUT, STATS>(out, 64, row0, rows, bias, acc, m, hi, bls);
  }
  if (STATS) {
    __syncthreads();
    for (int j = threadIdx.x; j < 128; j += blockDim.x) atomicAdd(&sums[j], bls[j]);
  }
}

// ---------------------------------------------------------------------------
// Node GEMM 1: nt = [h | agg] (N x 128) @ nw1 + nb1, fused stats
// ---------------------------------------------------------------------------
__global__ void k_node_gemm1(const float* __restrict__ h, const float* __restrict__ agg,
                             const float* __restrict__ W, const float* __restrict__ bias,
                             float* __restrict__ out, float* __restrict__ sums, int N)
{
  __shared__ _Float16 Bs[4 * 4 * 512];   // 16 KB
  __shared__ float bls[128];
  load_B_swz(Bs, W, 128, 128, 4);
  for (int j = threadIdx.x; j < 128; j += blockDim.x) bls[j] = 0.f;
  __syncthreads();
  const int lane = threadIdx.x & 31, wave = threadIdx.x >> 5;
  const int m = lane & 15, hi = lane >> 4;
  const int numTiles = (N + 15) >> 4;
  const int stride = gridDim.x * 8;
  for (int tile = blockIdx.x * 8 + wave; tile < numTiles; tile += stride) {
    const int row0 = tile * 16;
    const int row = min(row0 + m, N - 1);
    const float* hr = h   + (size_t)row * 64;
    const float* ar = agg + (size_t)row * 64;
    v8f acc[4] = {};
#pragma unroll
    for (int kc = 0; kc < 4; ++kc) {
      v16h a = frag_a_raw((kc < 2) ? hr : ar, (kc & 1)*32, hi);
#pragma unroll
      for (int n = 0; n < 4; ++n)
        acc[n] = wmma_f16(a, frag_b(Bs, kc*4 + n, lane), acc[n]);
    }
    store_tile<4, 0, true>(out, 64, row0, N, bias, acc, m, hi, bls);
  }
  __syncthreads();
  for (int j = threadIdx.x; j < 128; j += blockDim.x) atomicAdd(&sums[j], bls[j]);
}

// ---------------------------------------------------------------------------
// Readout: s_codes = h (N x 64) @ w_ro (64 x 32) + b_ro
// ---------------------------------------------------------------------------
__global__ void k_readout(const float* __restrict__ h, const float* __restrict__ W,
                          const float* __restrict__ bias, float* __restrict__ out, int N)
{
  __shared__ _Float16 Bs[2 * 2 * 512];   // 4 KB
  load_B_swz(Bs, W, 64, 64, 2);
  __syncthreads();
  const int lane = threadIdx.x & 31, wave = threadIdx.x >> 5;
  const int m = lane & 15, hi = lane >> 4;
  v16h bf[4];
#pragma unroll
  for (int f = 0; f < 4; ++f) bf[f] = frag_b(Bs, f, lane);
  const int numTiles = (N + 15) >> 4;
  const int stride = gridDim.x * 8;
  for (int tile = blockIdx.x * 8 + wave; tile < numTiles; tile += stride) {
    const int row0 = tile * 16;
    const float* rp = h + (size_t)min(row0 + m, N - 1) * 64;
    v8f acc[2] = {};
#pragma unroll
    for (int kc = 0; kc < 2; ++kc) {
      v16h a = frag_a_raw(rp, kc*32, hi);
#pragma unroll
      for (int n = 0; n < 2; ++n)
        acc[n] = wmma_f16(a, bf[kc*2 + n], acc[n]);
    }
    store_tile<2, 0, false>(out, 32, row0, N, bias, acc, m, hi, nullptr);
  }
}

__global__ void k_finalize_stats(const float* __restrict__ sums, int rows,
                                 const float* __restrict__ g, const float* __restrict__ bt,
                                 float* __restrict__ sc, float* __restrict__ sh)
{
  int j = threadIdx.x;
  if (j >= 64) return;
  float inv = 1.f / (float)rows;
  float mu  = sums[j] * inv;
  float var = sums[64 + j] * inv - mu * mu;
  float rs  = rsqrtf(var + 1e-5f);
  float s   = g[j] * rs;
  sc[j] = s;
  sh[j] = bt[j] - mu * s;
}

// ---------------------------------------------------------------------------
// Per-edge: vm = silu(norm(vt)) . vw2 ; scatter vd_norm*vm -> dx, ef -> agg
// ---------------------------------------------------------------------------
__global__ void k_vm_scatter(const float* __restrict__ vt, const float* __restrict__ sc,
                             const float* __restrict__ sh, const float* __restrict__ vw2,
                             const float* __restrict__ ef, const float* __restrict__ x,
                             const int* __restrict__ src, const int* __restrict__ dst,
                             float* __restrict__ agg, float* __restrict__ dx, int E)
{
  int e = blockIdx.x * blockDim.x + threadIdx.x;
  if (e >= E) return;
  const v4f* vt4 = (const v4f*)(vt + (size_t)e * 64);
  const v4f* sc4 = (const v4f*)sc;
  const v4f* sh4 = (const v4f*)sh;
  const v4f* w4  = (const v4f*)vw2;
  float vm = 0.f;
#pragma unroll
  for (int j = 0; j < 16; ++j) {
    v4f v = vt4[j] * sc4[j] + sh4[j];
    v4f w = w4[j];
#pragma unroll
    for (int c = 0; c < 4; ++c) vm += silu_f(v[c]) * w[c];
  }
  int s = src[e], d = dst[e];
  float vdx = x[3*s+0]-x[3*d+0], vdy = x[3*s+1]-x[3*d+1], vdz = x[3*s+2]-x[3*d+2];
  float inv = rsqrtf(vdx*vdx + vdy*vdy + vdz*vdz);
  atomicAdd(&dx[3*s+0], (vdx*inv + EPS8) * vm);
  atomicAdd(&dx[3*s+1], (vdy*inv + EPS8) * vm);
  atomicAdd(&dx[3*s+2], (vdz*inv + EPS8) * vm);
  const v4f* ef4 = (const v4f*)(ef + (size_t)e * 64);
  float* ag = agg + (size_t)s * 64;
#pragma unroll
  for (int j = 0; j < 16; ++j) {
    v4f v = ef4[j];
#pragma unroll
    for (int c = 0; c < 4; ++c) atomicAdd(&ag[j*4 + c], v[c]);
  }
}

__global__ void k_apply_dx(float* __restrict__ x, const float* __restrict__ dx, int n)
{
  int i = blockIdx.x * blockDim.x + threadIdx.x;
  if (i < n) x[i] += dx[i];
}

__global__ void k_copy(const float* __restrict__ a, float* __restrict__ b, int n)
{
  int i = blockIdx.x * blockDim.x + threadIdx.x;
  if (i < n) b[i] = a[i];
}

// ---------------------------------------------------------------------------
extern "C" void kernel_launch(void* const* d_in, const int* in_sizes, int n_in,
                              void* d_out, int out_size, void* d_ws, size_t ws_size,
                              hipStream_t stream)
{
  const float* pc     = (const float*)d_in[0];
  const float* w_emb  = (const float*)d_in[1];
  const float* b_emb  = (const float*)d_in[2];
  const float* ew1    = (const float*)d_in[3];
  const float* eb1    = (const float*)d_in[4];
  const float* eg     = (const float*)d_in[5];
  const float* ebt    = (const float*)d_in[6];
  const float* ew2    = (const float*)d_in[7];
  const float* eb2    = (const float*)d_in[8];
  const float* nw1    = (const float*)d_in[9];
  const float* nb1    = (const float*)d_in[10];
  const float* ng     = (const float*)d_in[11];
  const float* nbt    = (const float*)d_in[12];
  const float* nw2    = (const float*)d_in[13];
  const float* nb2    = (const float*)d_in[14];
  const float* vw1    = (const float*)d_in[15];
  const float* vb1    = (const float*)d_in[16];
  const float* vg     = (const float*)d_in[17];
  const float* vbt    = (const float*)d_in[18];
  const float* vw2    = (const float*)d_in[19];
  const float* w_ro   = (const float*)d_in[20];
  const float* b_ro   = (const float*)d_in[21];
  const int*   tag    = (const int*)d_in[22];
  const int*   edges  = (const int*)d_in[23];

  const int N = in_sizes[22];
  const int E = in_sizes[23] / 2;
  const int L = in_sizes[4] / 64;
  const int* src = edges;
  const int* dst = edges + E;

  // workspace carve-up, every region rounded to 32B so vector loads stay aligned
  float* w = (float*)d_ws;
  auto carve = [&w](size_t n) { float* p = w; w += (n + 7) & ~(size_t)7; return p; };
  float* barysum = carve(768);
  float* cnt     = carve(256);
  float* deg     = carve(N);
  float* densSum = carve(N);
  float* x       = carve((size_t)3 * N);
  float* dx      = carve((size_t)3 * N);
  float* h       = carve((size_t)64 * N);
  float* agg     = carve((size_t)64 * N);
  float* nt      = carve((size_t)64 * N);
  float* dir     = carve((size_t)3 * E);
  float* distb   = carve(E);
  float* ang     = carve(E);
  float* t1      = carve((size_t)64 * E);   // edge pre-norm buffer (reused for vt)
  float* ef      = carve((size_t)64 * E);
  float* sums    = carve(128);
  float* sc      = carve(64);
  float* sh      = carve(64);

  const int TB = 256;
  const int nB = (N + TB - 1) / TB;
  const int eB = (E + TB - 1) / TB;
  const int eTilesB = ((E + 15) / 16 + 7) / 8;
  const int nTilesB = ((N + 15) / 16 + 7) / 8;
  const int eGemmB = eTilesB < 640 ? eTilesB : 640;   // persistent: ~8+ tiles/wave
  const int nGemmB = nTilesB < 640 ? nTilesB : 640;

  hipMemsetAsync(barysum, 0, (size_t)(768 + 256 + 2 * (size_t)N + 16) * sizeof(float), stream);

  k_node_accum<<<nB, TB, 0, stream>>>(pc, tag, barysum, cnt, N);
  k_edge_geom<<<eB, TB, 0, stream>>>(pc, src, dst, dir, distb, deg, densSum, E);
  k_node_init<<<nB, TB, 0, stream>>>(pc, tag, barysum, cnt, deg, densSum, w_emb, b_emb, x, h, N);
  k_edge_attr<<<eB, TB, 0, stream>>>(dir, src, dst, ang, E);

  for (int l = 0; l < L; ++l) {
    const float* ew1l = ew1 + (size_t)l * 131 * 64;
    const float* eb1l = eb1 + (size_t)l * 64;
    const float* egl  = eg  + (size_t)l * 64;
    const float* ebtl = ebt + (size_t)l * 64;
    const float* ew2l = ew2 + (size_t)l * 64 * 64;
    const float* eb2l = eb2 + (size_t)l * 64;
    const float* nw1l = nw1 + (size_t)l * 128 * 64;
    const float* nb1l = nb1 + (size_t)l * 64;
    const float* ngl  = ng  + (size_t)l * 64;
    const float* nbtl = nbt + (size_t)l * 64;
    const float* nw2l = nw2 + (size_t)l * 64 * 64;
    const float* nb2l = nb2 + (size_t)l * 64;
    const float* vw1l = vw1 + (size_t)l * 64 * 64;
    const float* vb1l = vb1 + (size_t)l * 64;
    const float* vgl  = vg  + (size_t)l * 64;
    const float* vbtl = vbt + (size_t)l * 64;
    const float* vw2l = vw2 + (size_t)l * 64;

    // edge MLP layer 1, stats fused
    hipMemsetAsync(sums, 0, 128 * sizeof(float), stream);
    k_edge_gemm1<<<eGemmB, TB, 0, stream>>>(h, x, distb, ang, src, dst, ew1l, eb1l, t1, sums, E);
    k_finalize_stats<<<1, 64, 0, stream>>>(sums, E, egl, ebtl, sc, sh);

    // ef = silu( silu(norm(t1)) @ ew2 + eb2 )
    k_gemm64<1, 1, false><<<eGemmB, TB, 0, stream>>>(t1, sc, sh, ew2l, eb2l, ef, nullptr, E);

    // vt = ef @ vw1 + vb1 (into t1), stats fused (vg/vbt)
    hipMemsetAsync(sums, 0, 128 * sizeof(float), stream);
    k_gemm64<0, 0, true><<<eGemmB, TB, 0, stream>>>(ef, sc, sh, vw1l, vb1l, t1, sums, E);
    k_finalize_stats<<<1, 64, 0, stream>>>(sums, E, vgl, vbtl, sc, sh);

    // scatter: dx += vd_norm * vm ; agg += ef
    hipMemsetAsync(agg, 0, (size_t)64 * N * sizeof(float), stream);
    hipMemsetAsync(dx, 0, (size_t)3 * N * sizeof(float), stream);
    k_vm_scatter<<<eB, TB, 0, stream>>>(t1, sc, sh, vw2l, ef, x, src, dst, agg, dx, E);
    k_apply_dx<<<(3 * N + TB - 1) / TB, TB, 0, stream>>>(x, dx, 3 * N);

    // node MLP: nt = [h|agg] @ nw1 + nb1 (stats fused); h += silu(norm(nt)) @ nw2 + nb2
    hipMemsetAsync(sums, 0, 128 * sizeof(float), stream);
    k_node_gemm1<<<nGemmB, TB, 0, stream>>>(h, agg, nw1l, nb1l, nt, sums, N);
    k_finalize_stats<<<1, 64, 0, stream>>>(sums, N, ngl, nbtl, sc, sh);
    k_gemm64<1, 2, false><<<nGemmB, TB, 0, stream>>>(nt, sc, sh, nw2l, nb2l, h, nullptr, N);
  }

  // outputs: v_codes (3N floats) then s_codes (32N floats)
  float* out = (float*)d_out;
  k_copy<<<(3 * N + TB - 1) / TB, TB, 0, stream>>>(x, out, 3 * N);
  k_readout<<<nGemmB, TB, 0, stream>>>(h, w_ro, b_ro, out + (size_t)3 * N, N);
}
